// Quantizer_impl_3015067042299
// MI455X (gfx1250) — matrix operand order repaired and checked by reading, not compiled, not run on hardware
//
#include <hip/hip_runtime.h>
#include <stdint.h>

#define TOKENS   55296
#define SPATIAL  13824          // 24*24*24
#define CDIM     256
#define NEMBED   2048
#define QELEMS   (TOKENS*CDIM)  // 14155776
#define VQ_EPS   1e-5f

// ---- output layout (floats, concatenated in reference return order) ----
#define OFF_Q     0
#define OFF_LOSS  (QELEMS)                   // 14155776
#define OFF_IDX   (OFF_LOSS + 1)             // 14155777
#define OFF_NW    (OFF_IDX + TOKENS)         // 14211073
#define OFF_NN    (OFF_NW + NEMBED*CDIM)     // 14735361
#define OFF_NA    (OFF_NN + NEMBED)          // 14737409

// ---- workspace layout (bytes) ----
#define WS_WBF   0            // bf16 codebook: 2048*256*2 = 1048576
#define WS_WSQ   1048576      // f32 ||w||^2:   2048*4
#define WS_IDX   1056768      // int idx:       55296*4
#define WS_CNT   1277952      // f32 counts:    2048*4
#define WS_DW    1286144      // f32 dw:        2048*256*4
#define WS_LOSS  3383296      // f32 scalar
#define WS_NTOT  3383300      // f32 scalar
#define WS_ZERO_FLOATS 526338 // counts + dw + loss + ntot (contiguous)

typedef __attribute__((ext_vector_type(16))) __bf16 v16bf;
typedef __attribute__((ext_vector_type(8)))  float  v8f;

union Frag16 { v16bf v; uint32_t u[8]; uint4 q[2]; };
union Acc8   { v8f   v; float    f[8]; };

__device__ __forceinline__ uint16_t f2bf(float f) {
  uint32_t u = __float_as_uint(f);
  u += 0x7FFFu + ((u >> 16) & 1u);   // round-to-nearest-even
  return (uint16_t)(u >> 16);
}
__device__ __forceinline__ uint32_t pack_bf(float a, float b) {
  return (uint32_t)f2bf(a) | ((uint32_t)f2bf(b) << 16);
}

// ---------------------------------------------------------------- zero scratch
__global__ __launch_bounds__(256) void vq_zero(float* __restrict__ p, int n) {
  int i = blockIdx.x * 256 + threadIdx.x;
  if (i < n) p[i] = 0.0f;
}

// ------------------------------------------- codebook -> bf16 + exact ||w||^2
__global__ __launch_bounds__(256) void vq_prep_weight(
    const float* __restrict__ W, uint16_t* __restrict__ Wbf,
    float* __restrict__ Wsq) {
  __shared__ float red[256];
  const int r = blockIdx.x, c = threadIdx.x;
  float wv = W[r * CDIM + c];
  Wbf[r * CDIM + c] = f2bf(wv);
  red[c] = wv * wv;
  __syncthreads();
  for (int st = 128; st > 0; st >>= 1) {
    if (c < st) red[c] += red[c + st];
    __syncthreads();
  }
  if (c == 0) Wsq[r] = red[0];
}

// --------------------------------------------- WMMA nearest-codebook search
// block = 128 threads (4 waves), 64 tokens/block, all 2048 codes.
// Double-buffered LDS tile staging: global loads for tile nt+1 issue before
// the WMMA sequence of tile nt; one barrier per iteration.
#define BPAD 264   // padded bf16 row stride (528B = 33*16) -> conflict-free b128

__global__ __launch_bounds__(128) void vq_argmin_wmma(
    const float* __restrict__ X, const uint16_t* __restrict__ Wbf,
    const float* __restrict__ Wsq, float* __restrict__ out,
    int* __restrict__ idxOut) {
  __shared__ __align__(16) uint16_t sB[2][16 * BPAD];
  __shared__ float    sWsq[2][16];
  __shared__ float    sMV[64 * 16];
  __shared__ int      sMI[64 * 16];

  const int tid  = threadIdx.x;
  const int lane = tid & 31;
  const int wv   = tid >> 5;
  const int b    = blockIdx.x / 216;          // 216 blocks per batch image
  const int s0   = (blockIdx.x % 216) * 64;

  const int m  = lane & 15;                   // A row id / B-D column id
  const int hi = lane >> 4;
  const int tok = s0 + wv * 16 + m;           // this lane's A-matrix token

  // Preload A fragments (16 tokens x K=256, bf16) directly from global.
  // Lanes 0-15 read 16 consecutive tokens per channel -> coalesced 64B
  // segments. ISA 16x16x32 A layout: lanes 0-15 K=0-7,16-23; lanes 16-31
  // K=8-15,24-31 per K-step.
  Frag16 a[8];
  #pragma unroll
  for (int kt = 0; kt < 8; ++kt) {
    #pragma unroll
    for (int j = 0; j < 8; ++j) {
      int c0 = kt * 32 + hi * 8 + ((j < 4) ? (2 * j) : (16 + 2 * (j - 4)));
      const float* p = X + ((size_t)(b * CDIM + c0)) * SPATIAL + tok;
      a[kt].u[j] = pack_bf(p[0], p[SPATIAL]);
    }
  }

  float minv[8];
  int   mini[8];
  #pragma unroll
  for (int r = 0; r < 8; ++r) { minv[r] = 3.4e38f; mini[r] = 0; }

  // --- software pipeline: stage registers for the next 16-code tile --------
  uint4 stg[4];
  float wstg = 0.0f;
  const uint4* WbfQ = (const uint4*)Wbf;      // 32 uint4 per 256-ch code row

  // prologue: fetch + store tile 0
  #pragma unroll
  for (int it = 0; it < 4; ++it)
    stg[it] = WbfQ[(size_t)0 * 512 + it * 128 + tid];
  if (tid < 16) wstg = Wsq[tid];
  #pragma unroll
  for (int it = 0; it < 4; ++it) {
    int q = it * 128 + tid;                   // uint4 id 0..511
    *(uint4*)&sB[0][(q >> 5) * BPAD + (q & 31) * 8] = stg[it];
  }
  if (tid < 16) sWsq[0][tid] = wstg;
  __syncthreads();

  const v8f vzero = {};
  for (int nt = 0; nt < NEMBED / 16; ++nt) {
    const int cur = nt & 1;

    // issue global loads for tile nt+1 (latency hidden behind 8 WMMAs)
    if (nt + 1 < NEMBED / 16) {
      #pragma unroll
      for (int it = 0; it < 4; ++it)
        stg[it] = WbfQ[(size_t)(nt + 1) * 512 + it * 128 + tid];
      if (tid < 16) wstg = Wsq[(nt + 1) * 16 + tid];
    }

    Acc8 acc; acc.v = vzero;
    #pragma unroll
    for (int kt = 0; kt < 8; ++kt) {
      Frag16 bf;   // B: lane holds column n=m, K = kt*32 + hi*8 .. +7, kt*32 + 16 + hi*8 .. +7
      const uint4* p = (const uint4*)&sB[cur][m * BPAD + kt * 32 + hi * 8];
      bf.q[0] = p[0];
      bf.q[1] = p[2];
      acc.v = __builtin_amdgcn_wmma_f32_16x16x32_bf16(
          false, a[kt].v, false, bf.v, (short)0, acc.v, false, false);
    }

    const float wn  = sWsq[cur][m];
    const int  code = nt * 16 + m;
    #pragma unroll
    for (int r = 0; r < 8; ++r) {        // D: vgpr r -> token hi*8 + r
      float sc = wn - 2.0f * acc.f[r];   // ||w||^2 - 2 x.w  (||x||^2 const)
      if (sc < minv[r]) { minv[r] = sc; mini[r] = code; }
    }

    // store next tile into the other buffer (nobody reads it: the barrier
    // ending iteration nt-1 retired all reads of buffer 1-cur)
    if (nt + 1 < NEMBED / 16) {
      #pragma unroll
      for (int it = 0; it < 4; ++it) {
        int q = it * 128 + tid;
        *(uint4*)&sB[1 - cur][(q >> 5) * BPAD + (q & 31) * 8] = stg[it];
      }
      if (tid < 16) sWsq[1 - cur][tid] = wstg;
    }
    __syncthreads();
  }

  #pragma unroll
  for (int r = 0; r < 8; ++r) {
    int tl = wv * 16 + hi * 8 + r;       // token-local 0..63
    sMV[tl * 16 + m] = minv[r];
    sMI[tl * 16 + m] = mini[r];
  }
  __syncthreads();
  if (tid < 64) {                         // reduce 16 column-residues / token
    float bv = sMV[tid * 16];
    int   bi = sMI[tid * 16];
    #pragma unroll
    for (int j = 1; j < 16; ++j) {
      float v = sMV[tid * 16 + j];
      if (v < bv) { bv = v; bi = sMI[tid * 16 + j]; }
    }
    int t = b * SPATIAL + s0 + tid;
    idxOut[t] = bi;
    out[OFF_IDX + t] = (float)bi;
  }
}

// ---------------- gather codebook, emit quantized_st, loss, dw scatter ------
__global__ __launch_bounds__(256) void vq_quantize_scatter(
    const float* __restrict__ X, const float* __restrict__ W,
    const int* __restrict__ idx, float* __restrict__ out,
    float* __restrict__ dw, float* __restrict__ lossAcc) {
  __shared__ float red[256];
  size_t g = (size_t)blockIdx.x * 256 + threadIdx.x;
  int s = (int)(g % SPATIAL);
  int c = (int)((g / SPATIAL) % CDIM);
  int b = (int)(g / ((size_t)SPATIAL * CDIM));
  int t = b * SPATIAL + s;
  int code = idx[t];
  float xv = X[g];
  float qv = W[(size_t)code * CDIM + c];  // gather, L2-resident
  out[OFF_Q + g] = qv;                    // quantized_st == quantized (values)
  float d = qv - xv;
  atomicAdd(&dw[(size_t)code * CDIM + c], xv);
  red[threadIdx.x] = d * d;
  __syncthreads();
  for (int st = 128; st > 0; st >>= 1) {
    if (threadIdx.x < st) red[threadIdx.x] += red[threadIdx.x + st];
    __syncthreads();
  }
  if (threadIdx.x == 0) atomicAdd(lossAcc, red[0]);
}

// ------------------------------------------------------------- code counts
__global__ __launch_bounds__(256) void vq_counts(
    const int* __restrict__ idx, float* __restrict__ cnt) {
  int t = blockIdx.x * 256 + threadIdx.x;   // 216*256 == TOKENS exactly
  atomicAdd(&cnt[idx[t]], 1.0f);
}

// ------------------------------------ new_N, n_tot, finalize loss (1 block)
__global__ __launch_bounds__(256) void vq_finalize_n(
    const float* __restrict__ Nin, const float* __restrict__ cnt,
    const float* __restrict__ decayP, const float* __restrict__ ccP,
    const float* __restrict__ lossAcc, float* __restrict__ out,
    float* __restrict__ ntot) {
  __shared__ float red[256];
  const float decay = decayP[0];
  float local = 0.0f;
  for (int k = 0; k < NEMBED / 256; ++k) {
    int j = k * 256 + threadIdx.x;
    float nn = Nin[j] * decay + (1.0f - decay) * cnt[j];
    out[OFF_NN + j] = nn;
    local += nn;
  }
  red[threadIdx.x] = local;
  __syncthreads();
  for (int st = 128; st > 0; st >>= 1) {
    if (threadIdx.x < st) red[threadIdx.x] += red[threadIdx.x + st];
    __syncthreads();
  }
  if (threadIdx.x == 0) {
    ntot[0] = red[0];
    out[OFF_LOSS] = lossAcc[0] * ccP[0] / (float)QELEMS;
  }
}

// --------------------------------------- new_embed_avg / new_weight (EMA)
__global__ __launch_bounds__(256) void vq_finalize_w(
    const float* __restrict__ Avg, const float* __restrict__ dw,
    const float* __restrict__ decayP, const float* __restrict__ ntot,
    float* __restrict__ out) {
  const int j = blockIdx.x, c = threadIdx.x;
  const float decay = decayP[0];
  float e = Avg[(size_t)j * CDIM + c] * decay +
            (1.0f - decay) * dw[(size_t)j * CDIM + c];
  out[OFF_NA + (size_t)j * CDIM + c] = e;
  float nn = out[OFF_NN + j];
  float nt = ntot[0];
  float Wj = (nn + VQ_EPS) / (nt + (float)NEMBED * VQ_EPS) * nt;
  out[OFF_NW + (size_t)j * CDIM + c] = e / Wj;
}

extern "C" void kernel_launch(void* const* d_in, const int* in_sizes, int n_in,
                              void* d_out, int out_size, void* d_ws, size_t ws_size,
                              hipStream_t stream) {
  (void)in_sizes; (void)n_in; (void)out_size; (void)ws_size;
  const float* X     = (const float*)d_in[0];   // [4,256,24,24,24]
  const float* W     = (const float*)d_in[1];   // [2048,256]
  const float* Nin   = (const float*)d_in[2];   // [2048]
  const float* Avg   = (const float*)d_in[3];   // [2048,256]
  const float* decay = (const float*)d_in[4];   // scalar
  const float* cc    = (const float*)d_in[5];   // scalar
  float* out = (float*)d_out;
  char*  ws  = (char*)d_ws;

  uint16_t* Wbf  = (uint16_t*)(ws + WS_WBF);
  float* Wsq     = (float*)(ws + WS_WSQ);
  int*   idx     = (int*)(ws + WS_IDX);
  float* cnt     = (float*)(ws + WS_CNT);
  float* dw      = (float*)(ws + WS_DW);
  float* lossAcc = (float*)(ws + WS_LOSS);
  float* ntot    = (float*)(ws + WS_NTOT);

  vq_zero<<<(WS_ZERO_FLOATS + 255) / 256, 256, 0, stream>>>(
      (float*)(ws + WS_CNT), WS_ZERO_FLOATS);
  vq_prep_weight<<<NEMBED, 256, 0, stream>>>(W, Wbf, Wsq);
  vq_argmin_wmma<<<TOKENS / 64, 128, 0, stream>>>(X, Wbf, Wsq, out, idx);
  vq_quantize_scatter<<<QELEMS / 256, 256, 0, stream>>>(X, W, idx, out, dw, lossAcc);
  vq_counts<<<TOKENS / 256, 256, 0, stream>>>(idx, cnt);
  vq_finalize_n<<<1, 256, 0, stream>>>(Nin, cnt, decay, cc, lossAcc, out, ntot);
  vq_finalize_w<<<NEMBED, 256, 0, stream>>>(Avg, dw, decay, ntot, out);
}
